// transformerBlock_39384850104574
// MI455X (gfx1250) — matrix operand order repaired and checked by reading, not compiled, not run on hardware
//
#include <hip/hip_runtime.h>
#include <hip/hip_bf16.h>

typedef __bf16 bf16;
typedef __attribute__((ext_vector_type(16))) __bf16 v16bf;
typedef __attribute__((ext_vector_type(8)))  float   v8f;
typedef __attribute__((ext_vector_type(4)))  __bf16  bf16x4;

union Frag {
    v16bf v;
    uint4 u[2];
};

__device__ __forceinline__ v8f wmma_bf16(const Frag& a, const Frag& b, v8f c) {
    return __builtin_amdgcn_wmma_f32_16x16x32_bf16(false, a.v, false, b.v,
                                                   (short)0, c, false, false);
}

// ---------------------------------------------------------------------------
// Weight transpose fp32[K][N] -> bf16[N][K]
// ---------------------------------------------------------------------------
__global__ __launch_bounds__(256) void transpose_bf16_kernel(
    const float* __restrict__ w, bf16* __restrict__ wt, int K, int N)
{
    __shared__ float t[32][33];
    const int nb = blockIdx.x * 32, kb = blockIdx.y * 32;
    const int tx = threadIdx.x, ty = threadIdx.y;   // block (32, 8)
#pragma unroll
    for (int j = 0; j < 4; ++j) {
        int kk = kb + ty + j * 8;
        t[ty + j * 8][tx] = w[(size_t)kk * N + nb + tx];
    }
    __syncthreads();
#pragma unroll
    for (int j = 0; j < 4; ++j) {
        int nn = nb + ty + j * 8;
        wt[(size_t)nn * K + kb + tx] = (bf16)t[tx][ty + j * 8];
    }
}

// ---------------------------------------------------------------------------
// LayerNorm over D=1024. Optional residual gather from obuf [B,H,N,C].
// Writes bf16 normalized row; optionally writes fp32 (x + o) row.
// ---------------------------------------------------------------------------
__global__ __launch_bounds__(256) void ln_kernel(
    const float* __restrict__ x, const float* __restrict__ obuf,
    const float* __restrict__ g, const float* __restrict__ bb,
    float* __restrict__ yfull, bf16* __restrict__ yln)
{
    const int row = blockIdx.x;          // 0..8191 = b*2048 + n
    const int tid = threadIdx.x;
    const int d0 = tid * 4;
    float4 v = *(const float4*)(x + (size_t)row * 1024 + d0);
    if (obuf) {
        int b_ = row >> 11, n = row & 2047;
        int h = d0 >> 7, c = d0 & 127;
        float4 ov = *(const float4*)(obuf + (((size_t)(b_ * 8 + h) * 2048 + n) * 128 + c));
        v.x += ov.x; v.y += ov.y; v.z += ov.z; v.w += ov.w;
    }
    if (yfull) *(float4*)(yfull + (size_t)row * 1024 + d0) = v;

    float s = v.x + v.y + v.z + v.w;
    float q = v.x * v.x + v.y * v.y + v.z * v.z + v.w * v.w;
#pragma unroll
    for (int m = 1; m < 32; m <<= 1) { s += __shfl_xor(s, m); q += __shfl_xor(q, m); }
    __shared__ float ps[8], pq[8];
    const int w = tid >> 5, lane = tid & 31;
    if (lane == 0) { ps[w] = s; pq[w] = q; }
    __syncthreads();
    float S = 0.f, Q = 0.f;
#pragma unroll
    for (int i = 0; i < 8; ++i) { S += ps[i]; Q += pq[i]; }
    const float mean = S * (1.0f / 1024.0f);
    const float var  = Q * (1.0f / 1024.0f) - mean * mean;
    const float rs   = rsqrtf(var + 1e-3f);

    bf16x4 o;
    o[0] = (bf16)((v.x - mean) * rs * g[d0 + 0] + bb[d0 + 0]);
    o[1] = (bf16)((v.y - mean) * rs * g[d0 + 1] + bb[d0 + 1]);
    o[2] = (bf16)((v.z - mean) * rs * g[d0 + 2] + bb[d0 + 2]);
    o[3] = (bf16)((v.w - mean) * rs * g[d0 + 3] + bb[d0 + 3]);
    *(bf16x4*)(yln + (size_t)row * 1024 + d0) = o;
}

// ---------------------------------------------------------------------------
// WMMA bf16 GEMM: C[M,N] = A[M,K] @ Bt[N,K]^T (+ bias, epilogue by MODE)
// Block tile 128x128, 8 waves in 4(M)x2(N); wave tile 32x64 = 2x4 WMMA tiles.
// Double-buffered LDS with register staging: one barrier per 32-K step;
// next tile's global_load_b128s overlap the 8 WMMAs of the current step.
// MODE 0: relu, scatter QKV de-interleave -> q(*0.125), k, vT (bf16)
// MODE 1: relu -> hbuf bf16 [M][NN]
// MODE 2: + ybuf residual -> fp32 out [M][1024]
// ---------------------------------------------------------------------------
template <int MODE>
__global__ __launch_bounds__(256) void gemm_bf16_kernel(
    const bf16* __restrict__ A, const bf16* __restrict__ Bt,
    const float* __restrict__ bias, int K, int NN,
    bf16* __restrict__ qb, bf16* __restrict__ kbuf, bf16* __restrict__ vtb,
    bf16* __restrict__ hb, const float* __restrict__ ybuf, float* __restrict__ outb)
{
    __shared__ __attribute__((aligned(16))) bf16 lA[2][128 * 40];
    __shared__ __attribute__((aligned(16))) bf16 lB[2][128 * 40];

    const int tid = threadIdx.x;
    const int m0 = blockIdx.y * 128;
    const int n0 = blockIdx.x * 128;
    const int lane = tid & 31, w = tid >> 5;
    const int wm = (w >> 1) * 32;       // 0,32,64,96
    const int wn = (w & 1) * 64;        // 0,64
    const int lm = lane & 15, hi = lane >> 4;

    // staging: A tile 128x32 = 512 16B-chunks, B tile 128x32 = 512 chunks
    const int c0 = tid, c1 = tid + 256;
    const int ar0 = c0 >> 2, ac0 = (c0 & 3) * 8;
    const int ar1 = c1 >> 2, ac1 = (c1 & 3) * 8;
    const bf16* Ag0 = A  + (size_t)(m0 + ar0) * K + ac0;
    const bf16* Ag1 = A  + (size_t)(m0 + ar1) * K + ac1;
    const bf16* Bg0 = Bt + (size_t)(n0 + ar0) * K + ac0;
    const bf16* Bg1 = Bt + (size_t)(n0 + ar1) * K + ac1;

    const v8f vzero = {0.f, 0.f, 0.f, 0.f, 0.f, 0.f, 0.f, 0.f};
    v8f acc[2][4];
#pragma unroll
    for (int mi = 0; mi < 2; ++mi)
#pragma unroll
        for (int ni = 0; ni < 4; ++ni) acc[mi][ni] = vzero;

    const int nk = K >> 5;
    uint4 ra0 = *(const uint4*)Ag0;
    uint4 ra1 = *(const uint4*)Ag1;
    uint4 rb0 = *(const uint4*)Bg0;
    uint4 rb1 = *(const uint4*)Bg1;

    for (int kt = 0; kt < nk; ++kt) {
        const int cur = kt & 1;
        *(uint4*)&lA[cur][ar0 * 40 + ac0] = ra0;
        *(uint4*)&lA[cur][ar1 * 40 + ac1] = ra1;
        *(uint4*)&lB[cur][ar0 * 40 + ac0] = rb0;
        *(uint4*)&lB[cur][ar1 * 40 + ac1] = rb1;
        __syncthreads();

        if (kt + 1 < nk) {                       // prefetch next 32-K slice
            const int ko = (kt + 1) * 32;
            ra0 = *(const uint4*)(Ag0 + ko);
            ra1 = *(const uint4*)(Ag1 + ko);
            rb0 = *(const uint4*)(Bg0 + ko);
            rb1 = *(const uint4*)(Bg1 + ko);
        }

        Frag a[2], b[4];
#pragma unroll
        for (int mi = 0; mi < 2; ++mi) {
            const bf16* p = &lA[cur][(wm + mi * 16 + lm) * 40 + hi * 8];
            a[mi].u[0] = *(const uint4*)p;
            a[mi].u[1] = *(const uint4*)(p + 16);
        }
#pragma unroll
        for (int ni = 0; ni < 4; ++ni) {
            const bf16* p = &lB[cur][(wn + ni * 16 + lm) * 40 + hi * 8];
            b[ni].u[0] = *(const uint4*)p;
            b[ni].u[1] = *(const uint4*)(p + 16);
        }
#pragma unroll
        for (int mi = 0; mi < 2; ++mi)
#pragma unroll
            for (int ni = 0; ni < 4; ++ni)
                acc[mi][ni] = wmma_bf16(a[mi], b[ni], acc[mi][ni]);
        // no second barrier: LDS reads are consumed by WMMA before the next
        // iteration's barrier, and the next write targets the other buffer.
    }

#pragma unroll
    for (int mi = 0; mi < 2; ++mi)
#pragma unroll
        for (int ni = 0; ni < 4; ++ni)
#pragma unroll
            for (int r = 0; r < 8; ++r) {
                const int grow = m0 + wm + mi * 16 + r + 8 * hi;
                const int gcol = n0 + wn + ni * 16 + lm;
                float val = acc[mi][ni][r] + bias[gcol];
                if (MODE == 0) {
                    val = fmaxf(val, 0.0f);
                    // column j factors as (c,h,i): j = c*24 + h*3 + i
                    const int i3 = gcol % 3;
                    const int hh = (gcol / 3) & 7;
                    const int c  = gcol / 24;
                    const int b_ = grow >> 11, n = grow & 2047;
                    const size_t idx = ((size_t)(b_ * 8 + hh) * 2048 + n) * 128 + c;
                    if (i3 == 0)      qb[idx] = (bf16)(val * 0.125f);   // fold SCALE
                    else if (i3 == 1) kbuf[idx] = (bf16)val;
                    else vtb[((size_t)(b_ * 8 + hh) * 128 + c) * 2048 + n] = (bf16)val;
                } else if (MODE == 1) {
                    val = fmaxf(val, 0.0f);
                    hb[(size_t)grow * NN + gcol] = (bf16)val;
                } else {
                    val += ybuf[(size_t)grow * 1024 + gcol];
                    outb[(size_t)grow * 1024 + gcol] = val;
                }
            }
}

// ---------------------------------------------------------------------------
// Flash attention. One wave per 16-query tile. q,k: [pair][n][c] bf16 (q
// pre-scaled), vT: [pair][c][n] bf16. Output obuf fp32 [pair][n][c].
// ---------------------------------------------------------------------------
__global__ __launch_bounds__(256) void attn_kernel(
    const bf16* __restrict__ qg, const bf16* __restrict__ kg,
    const bf16* __restrict__ vtg, float* __restrict__ obuf)
{
    __shared__ __attribute__((aligned(16))) bf16 lP[8 * 16 * 40];

    const int tid = threadIdx.x;
    const int lane = tid & 31, w = tid >> 5;
    const int lm = lane & 15, hi = lane >> 4;
    const int gw = blockIdx.x * 8 + w;       // 0..4095
    const int pair = gw >> 7;                // b*8 + h
    const int qt = gw & 127;
    const size_t base = (size_t)pair * 2048 * 128;

    // Q fragments for all 4 K-steps of C=128 (kept in registers)
    Frag qf[4];
#pragma unroll
    for (int cs = 0; cs < 4; ++cs) {
        const bf16* p = qg + base + (size_t)(qt * 16 + lm) * 128 + cs * 32 + hi * 8;
        qf[cs].u[0] = *(const uint4*)p;
        qf[cs].u[1] = *(const uint4*)(p + 16);
    }

    const v8f vzero = {0.f, 0.f, 0.f, 0.f, 0.f, 0.f, 0.f, 0.f};
    v8f oacc[8];
#pragma unroll
    for (int g = 0; g < 8; ++g) oacc[g] = vzero;
    float mrun[8], lrun[8];
#pragma unroll
    for (int r = 0; r < 8; ++r) { mrun[r] = -1e30f; lrun[r] = 0.0f; }

    bf16* pw = lP + w * 640;     // private 16x40 bf16 tile per wave

    for (int kb0 = 0; kb0 < 2048; kb0 += 32) {
        v8f s0 = vzero, s1 = vzero;
#pragma unroll
        for (int cs = 0; cs < 4; ++cs) {
            Frag kf0, kf1;
            const bf16* kp0 = kg + base + (size_t)(kb0 + lm) * 128 + cs * 32 + hi * 8;
            kf0.u[0] = *(const uint4*)kp0;
            kf0.u[1] = *(const uint4*)(kp0 + 16);
            const bf16* kp1 = kp0 + 16 * 128;
            kf1.u[0] = *(const uint4*)kp1;
            kf1.u[1] = *(const uint4*)(kp1 + 16);
            s0 = wmma_bf16(qf[cs], kf0, s0);
            s1 = wmma_bf16(qf[cs], kf1, s1);
        }

        // online softmax: row stats live in accumulator index r across 16 lanes
        float alpha[8];
#pragma unroll
        for (int r = 0; r < 8; ++r) {
            float sm = fmaxf(s0[r], s1[r]);
#pragma unroll
            for (int m = 1; m < 16; m <<= 1) sm = fmaxf(sm, __shfl_xor(sm, m));
            const float mn = fmaxf(mrun[r], sm);
            alpha[r] = __expf(mrun[r] - mn);
            const float p0 = __expf(s0[r] - mn);
            const float p1 = __expf(s1[r] - mn);
            s0[r] = p0; s1[r] = p1;
            float rsum = p0 + p1;
#pragma unroll
            for (int m = 1; m < 16; m <<= 1) rsum += __shfl_xor(rsum, m);
            lrun[r] = lrun[r] * alpha[r] + rsum;
            mrun[r] = mn;
        }
#pragma unroll
        for (int g = 0; g < 8; ++g)
#pragma unroll
            for (int r = 0; r < 8; ++r) oacc[g][r] *= alpha[r];

        // D-layout -> A-layout relayout of P through LDS (bf16)
#pragma unroll
        for (int r = 0; r < 8; ++r) {
            const int row = r + 8 * hi;
            pw[row * 40 + lm]      = (bf16)s0[r];
            pw[row * 40 + 16 + lm] = (bf16)s1[r];
        }
        asm volatile("s_wait_dscnt 0" ::: "memory");

        Frag pf;
        {
            const bf16* pp = pw + lm * 40 + hi * 8;
            pf.u[0] = *(const uint4*)pp;
            pf.u[1] = *(const uint4*)(pp + 16);
        }
#pragma unroll
        for (int g = 0; g < 8; ++g) {
            Frag vf;
            const bf16* vp = vtg + base + (size_t)(g * 16 + lm) * 2048 + kb0 + hi * 8;
            vf.u[0] = *(const uint4*)vp;
            vf.u[1] = *(const uint4*)(vp + 16);
            oacc[g] = wmma_bf16(pf, vf, oacc[g]);
        }
    }

#pragma unroll
    for (int r = 0; r < 8; ++r) {
        const float inv = 1.0f / lrun[r];
        const size_t row = (size_t)pair * 2048 + qt * 16 + r + 8 * hi;
#pragma unroll
        for (int g = 0; g < 8; ++g)
            obuf[row * 128 + g * 16 + lm] = oacc[g][r] * inv;
    }
}

// ---------------------------------------------------------------------------
// Launch
// ---------------------------------------------------------------------------
extern "C" void kernel_launch(void* const* d_in, const int* in_sizes, int n_in,
                              void* d_out, int out_size, void* d_ws, size_t ws_size,
                              hipStream_t stream) {
    (void)in_sizes; (void)n_in; (void)out_size; (void)ws_size;
    const float* x    = (const float*)d_in[0];
    const float* ln_g = (const float*)d_in[1];
    const float* ln_b = (const float*)d_in[2];
    const float* w3   = (const float*)d_in[3];
    const float* b3   = (const float*)d_in[4];
    const float* w1   = (const float*)d_in[5];
    const float* b1   = (const float*)d_in[6];
    const float* w2   = (const float*)d_in[7];
    const float* b2   = (const float*)d_in[8];
    float* out = (float*)d_out;

    char* ws = (char*)d_ws;
    size_t off = 0;
    auto alloc = [&](size_t bytes) -> char* {
        char* p = ws + off;
        off += (bytes + 255) & ~(size_t)255;
        return p;
    };
    const size_t MR = 8192;  // B*N rows
    bf16*  w3t  = (bf16*)alloc(3072 * 1024 * 2);
    bf16*  w1t  = (bf16*)alloc(4096 * 1024 * 2);
    bf16*  w2t  = (bf16*)alloc((size_t)1024 * 4096 * 2);
    bf16*  yln  = (bf16*)alloc(MR * 1024 * 2);
    bf16*  qb   = (bf16*)alloc(MR * 1024 * 2);
    bf16*  kb   = (bf16*)alloc(MR * 1024 * 2);
    bf16*  vtb  = (bf16*)alloc(MR * 1024 * 2);
    float* obuf = (float*)alloc(MR * 1024 * 4);
    float* ybuf = (float*)alloc(MR * 1024 * 4);
    bf16*  y1   = (bf16*)alloc(MR * 1024 * 2);
    bf16*  hb   = (bf16*)alloc(MR * 4096 * 2);

    dim3 tb(32, 8);
    transpose_bf16_kernel<<<dim3(3072 / 32, 1024 / 32), tb, 0, stream>>>(w3, w3t, 1024, 3072);
    transpose_bf16_kernel<<<dim3(4096 / 32, 1024 / 32), tb, 0, stream>>>(w1, w1t, 1024, 4096);
    transpose_bf16_kernel<<<dim3(1024 / 32, 4096 / 32), tb, 0, stream>>>(w2, w2t, 4096, 1024);

    ln_kernel<<<8192, 256, 0, stream>>>(x, nullptr, ln_g, ln_b, nullptr, yln);

    gemm_bf16_kernel<0><<<dim3(3072 / 128, 8192 / 128), 256, 0, stream>>>(
        yln, w3t, b3, 1024, 3072, qb, kb, vtb, nullptr, nullptr, nullptr);

    attn_kernel<<<512, 256, 0, stream>>>(qb, kb, vtb, obuf);

    ln_kernel<<<8192, 256, 0, stream>>>(x, obuf, ln_g, ln_b, ybuf, y1);

    gemm_bf16_kernel<1><<<dim3(4096 / 128, 8192 / 128), 256, 0, stream>>>(
        y1, w1t, b1, 1024, 4096, nullptr, nullptr, nullptr, hb, nullptr, nullptr);

    gemm_bf16_kernel<2><<<dim3(1024 / 128, 8192 / 128), 256, 0, stream>>>(
        hb, w2t, b2, 4096, 1024, nullptr, nullptr, nullptr, nullptr, ybuf, out);
}
